// MiniTransformer_59785944760582
// MI455X (gfx1250) — matrix operand order repaired and checked by reading, not compiled, not run on hardware
//
#include <hip/hip_runtime.h>
#include <hip/hip_bf16.h>
#include <math.h>

typedef __attribute__((ext_vector_type(16))) _Float16 v16h;
typedef __attribute__((ext_vector_type(8)))  _Float16 v8h;
typedef __attribute__((ext_vector_type(8)))  float    v8f;
typedef __attribute__((ext_vector_type(4)))  float    v4f;
typedef __attribute__((ext_vector_type(2)))  float    v2f;

#define XS 68    // X row stride (f32)
#define AS 136   // A-stage row stride (f16)
#define QS 200   // QKV row stride (f32)

struct WaveShm {
  float    X[16 * XS];     // master activation (residual stream), f32
  _Float16 A[16 * AS];     // WMMA A-matrix staging (LN output / ctx / yn), f16
  float    QK[16 * QS];    // QKV result f32; reused as f16 h1 staging for FFN2
};

// ---- packed-weight offsets (in halfs) in d_ws ----
#define WQ_OFF  0       // in_proj: 12 tiles x 2 K-chunks
#define WO_OFF  12288   // out_w  :  4 tiles x 2 K-chunks
#define W1_OFF  16384   // w1     :  8 tiles x 2 K-chunks
#define W2_OFF  24576   // w2     :  4 tiles x 4 K-chunks
#define WTOTAL  32768

// ------------------------------------------------------------------
// Pack fp32 torch-layout weights (out,in) into per-lane B-fragment
// order: [tile][chunk][lane][16 halfs]. B 32x16 f16 layout: lanes 0-15
// hold K {0-7,16-23}, lanes 16-31 hold K {8-15,24-31}; column = lane&15.
// ------------------------------------------------------------------
__global__ void pack_weights(const float* __restrict__ qkvw,
                             const float* __restrict__ outw,
                             const float* __restrict__ w1,
                             const float* __restrict__ w2,
                             _Float16* __restrict__ dst) {
  const int idx = blockIdx.x * blockDim.x + threadIdx.x;
  if (idx >= WTOTAL) return;
  const float* src; int chunks, K, local;
  if (idx < WO_OFF)      { src = qkvw; chunks = 2; K = 64;  local = idx; }
  else if (idx < W1_OFF) { src = outw; chunks = 2; K = 64;  local = idx - WO_OFF; }
  else if (idx < W2_OFF) { src = w1;   chunks = 2; K = 64;  local = idx - W1_OFF; }
  else                   { src = w2;   chunks = 4; K = 128; local = idx - W2_OFF; }
  const int h    = local & 15;
  const int lane = (local >> 4) & 31;
  const int fc   = local >> 9;
  const int chunk = fc % chunks;
  const int tile  = fc / chunks;
  const int e = tile * 16 + (lane & 15);            // output column
  const int k = chunk * 32 + (h & 7) + ((h >= 8) ? 16 : 0) + ((lane >= 16) ? 8 : 0);
  dst[idx] = (_Float16)src[(size_t)e * K + k];
}

// ------------------------------------------------------------------
// A-fragment loader (16-bit A 16x32 layout): lane<16 holds K {0-7,16-23},
// lane>=16 holds K {8-15,24-31} of row (lane&15).  2x ds_load_b128.
// ------------------------------------------------------------------
__device__ __forceinline__ v16h afrag(const _Float16* rowp, int c0, bool hi) {
  const int off = c0 + (hi ? 8 : 0);
  v8h lo = *(const v8h*)(rowp + off);
  v8h hh = *(const v8h*)(rowp + off + 16);
  v16h a;
#pragma unroll
  for (int i = 0; i < 8; ++i) { a[i] = lo[i]; a[i + 8] = hh[i]; }
  return a;
}

__device__ __forceinline__ v16h bfrag(const _Float16* pw, int f, int lane) {
  return *(const v16h*)(pw + ((size_t)(f * 32 + lane) << 4));  // 32B/lane, coalesced
}

__device__ __forceinline__ v8f wmma_f16(v16h a, v16h b, v8f c) {
  return __builtin_amdgcn_wmma_f32_16x16x32_f16(false, a, false, b, (short)0, c,
                                                false, false);
}

// LayerNorm over D=64: lanes (l, l+16) split a row's columns 0-31 / 32-63,
// combine via shfl_xor(16) (wave32), write f16 into A-stage.
__device__ __forceinline__ void layernorm_to_A(WaveShm& W, const float* g,
                                               const float* b, int lane) {
  const int row = lane & 15;
  const int cb  = (lane >= 16) ? 32 : 0;
  float x[32];
#pragma unroll
  for (int i = 0; i < 8; ++i) {
    v4f v = *(const v4f*)(&W.X[row * XS + cb + 4 * i]);
#pragma unroll
    for (int j = 0; j < 4; ++j) x[4 * i + j] = v[j];
  }
  float s1 = 0.f, s2 = 0.f;
#pragma unroll
  for (int i = 0; i < 32; ++i) { s1 += x[i]; s2 += x[i] * x[i]; }
  s1 += __shfl_xor(s1, 16, 32);
  s2 += __shfl_xor(s2, 16, 32);
  const float mu  = s1 * (1.0f / 64.0f);
  const float var = s2 * (1.0f / 64.0f) - mu * mu;
  const float rs  = rsqrtf(var + 1e-5f);
  _Float16 y[32];
#pragma unroll
  for (int i = 0; i < 8; ++i) {
    v4f gv = *(const v4f*)(g + cb + 4 * i);
    v4f bv = *(const v4f*)(b + cb + 4 * i);
#pragma unroll
    for (int j = 0; j < 4; ++j)
      y[4 * i + j] = (_Float16)((x[4 * i + j] - mu) * rs * gv[j] + bv[j]);
  }
#pragma unroll
  for (int i = 0; i < 4; ++i) {
    v8h v;
#pragma unroll
    for (int j = 0; j < 8; ++j) v[j] = y[8 * i + j];
    *(v8h*)(&W.A[row * AS + cb + 8 * i]) = v;
  }
}

// ------------------------------------------------------------------
// Fused mini-transformer: one wave = 4 token groups (16 rows).
// ------------------------------------------------------------------
__global__ __launch_bounds__(64) void mini_transformer_fused(
    const float* __restrict__ z_m, const float* __restrict__ z_c,
    const float* __restrict__ z_t, const float* __restrict__ z_cls,
    const float* __restrict__ in_proj_b, const float* __restrict__ out_b,
    const float* __restrict__ ln1_g, const float* __restrict__ ln1_b,
    const float* __restrict__ ln2_g, const float* __restrict__ ln2_b,
    const float* __restrict__ b1, const float* __restrict__ b2,
    const _Float16* __restrict__ wq, const _Float16* __restrict__ wo,
    const _Float16* __restrict__ w1p, const _Float16* __restrict__ w2p,
    float* __restrict__ out_h, float* __restrict__ out_aw, int Mtot) {
  __shared__ WaveShm shm[2];
  const int lane = threadIdx.x & 31;
  const int wv   = threadIdx.x >> 5;
  WaveShm& W     = shm[wv];
  const int G    = (blockIdx.x * 2 + wv) * 4;  // first token-group of this wave
  const int m_lo = lane & 15;
  const bool hi  = lane >= 16;
  const int mb   = hi ? 8 : 0;

  // Warm the 64KB packed-weight block (global_prefetch_b8).
#pragma unroll
  for (int i = 0; i < 8; ++i)
    __builtin_prefetch(((const char*)wq) + ((size_t)threadIdx.x * 8 + i) * 128, 0, 1);

  // ---- Stage 1: build X = [cls, z_m, z_c, z_t] rows (f32) ----
#pragma unroll
  for (int r = 0; r < 16; ++r) {
    int g = G + (r >> 2);
    if (g >= Mtot) g = Mtot - 1;
    const int s = r & 3;
    const float* src;
    if (s == 0) src = z_cls;
    else {
      const float* base = (s == 1) ? z_m : (s == 2) ? z_c : z_t;
      src = base + (size_t)g * 64;
    }
    *(v2f*)(&W.X[r * XS + 2 * lane]) = *(const v2f*)(src + 2 * lane);
  }
  __syncthreads();

  // ---- Stage 2: LN1 -> A-stage ----
  layernorm_to_A(W, ln1_g, ln1_b, lane);
  __syncthreads();

  // ---- Stage 3: QKV GEMM, 12 N-tiles x 2 K-chunks (24 WMMAs) ----
  {
    v16h a0 = afrag(&W.A[m_lo * AS], 0, hi);
    v16h a1 = afrag(&W.A[m_lo * AS], 32, hi);
#pragma unroll
    for (int t = 0; t < 12; ++t) {
      v8f acc = {};
      acc = wmma_f16(a0, bfrag(wq, t * 2 + 0, lane), acc);
      acc = wmma_f16(a1, bfrag(wq, t * 2 + 1, lane), acc);
      const float bias = in_proj_b[t * 16 + m_lo];
#pragma unroll
      for (int r = 0; r < 8; ++r)
        W.QK[(r + mb) * QS + t * 16 + m_lo] = acc[r] + bias;  // C-layout scatter
    }
  }
  __syncthreads();

  // ---- Stage 4: 4x4 attention per (row, head) on VALU; 2 heads/lane ----
  {
    const int m  = m_lo;
    const int bm = m & ~3;  // group base row
    float aw2[2][4];
    _Float16* Ast = &W.A[m * AS];
#pragma unroll
    for (int hh = 0; hh < 2; ++hh) {
      const int h  = (hi ? 1 : 0) + hh * 2;
      const int qo = h * 16, ko = 64 + h * 16, vo = 128 + h * 16;
      float q[16];
#pragma unroll
      for (int i = 0; i < 4; ++i) {
        v4f v = *(const v4f*)(&W.QK[m * QS + qo + 4 * i]);
        q[4 * i] = v[0]; q[4 * i + 1] = v[1]; q[4 * i + 2] = v[2]; q[4 * i + 3] = v[3];
      }
      float sc[4];
#pragma unroll
      for (int j = 0; j < 4; ++j) {
        float a = 0.f;
#pragma unroll
        for (int d = 0; d < 16; ++d) a += q[d] * W.QK[(bm + j) * QS + ko + d];
        sc[j] = a * 0.25f;  // 1/sqrt(16)
      }
      const float mx = fmaxf(fmaxf(sc[0], sc[1]), fmaxf(sc[2], sc[3]));
      float se = 0.f;
#pragma unroll
      for (int j = 0; j < 4; ++j) { sc[j] = __expf(sc[j] - mx); se += sc[j]; }
      const float inv = 1.0f / se;
#pragma unroll
      for (int j = 0; j < 4; ++j) { sc[j] *= inv; aw2[hh][j] = sc[j]; }
      _Float16 cx[16];
#pragma unroll
      for (int d = 0; d < 16; ++d) {
        float a = 0.f;
#pragma unroll
        for (int j = 0; j < 4; ++j) a += sc[j] * W.QK[(bm + j) * QS + vo + d];
        cx[d] = (_Float16)a;
      }
#pragma unroll
      for (int i = 0; i < 2; ++i) {
        v8h v;
#pragma unroll
        for (int j = 0; j < 8; ++j) v[j] = cx[8 * i + j];
        *(v8h*)(Ast + h * 16 + 8 * i) = v;  // ctx -> A-stage (f16)
      }
    }
    // head-averaged attention weights -> d_out (coalesced float4 per row)
    float wsm[4];
#pragma unroll
    for (int j = 0; j < 4; ++j) {
      float p = aw2[0][j] + aw2[1][j];
      p += __shfl_xor(p, 16, 32);
      wsm[j] = p * 0.25f;
    }
    const int g = G + (m >> 2);
    if (!hi && g < Mtot) {
      v4f v = {wsm[0], wsm[1], wsm[2], wsm[3]};
      *(v4f*)(out_aw + (size_t)g * 16 + (m & 3) * 4) = v;
    }
  }
  __syncthreads();

  // ---- Stage 5: out-proj (8 WMMAs) + residual via ds_add_f32 ----
  {
    v16h a0 = afrag(&W.A[m_lo * AS], 0, hi);
    v16h a1 = afrag(&W.A[m_lo * AS], 32, hi);
#pragma unroll
    for (int t = 0; t < 4; ++t) {
      v8f acc = {};
      acc = wmma_f16(a0, bfrag(wo, t * 2 + 0, lane), acc);
      acc = wmma_f16(a1, bfrag(wo, t * 2 + 1, lane), acc);
      const float bias = out_b[t * 16 + m_lo];
#pragma unroll
      for (int r = 0; r < 8; ++r)
        atomicAdd(&W.X[(r + mb) * XS + t * 16 + m_lo], acc[r] + bias);
    }
  }
  __syncthreads();

  // ---- Stage 6: LN2 -> A-stage ----
  layernorm_to_A(W, ln2_g, ln2_b, lane);
  __syncthreads();

  // ---- Stage 7: FFN1 (16 WMMAs) + exact GELU -> f16 H-stage ----
  _Float16* Hs = (_Float16*)W.QK;  // QKV buffer reused as f16 staging
  {
    v16h a0 = afrag(&W.A[m_lo * AS], 0, hi);
    v16h a1 = afrag(&W.A[m_lo * AS], 32, hi);
#pragma unroll
    for (int t = 0; t < 8; ++t) {
      v8f acc = {};
      acc = wmma_f16(a0, bfrag(w1p, t * 2 + 0, lane), acc);
      acc = wmma_f16(a1, bfrag(w1p, t * 2 + 1, lane), acc);
      const float bias = b1[t * 16 + m_lo];
#pragma unroll
      for (int r = 0; r < 8; ++r) {
        const float v  = acc[r] + bias;
        const float gl = 0.5f * v * (1.0f + erff(v * 0.70710678118654752f));
        Hs[(r + mb) * AS + t * 16 + m_lo] = (_Float16)gl;
      }
    }
  }
  __syncthreads();

  // ---- Stage 8: FFN2, K=128 -> 4 chunks (16 WMMAs) + residual ----
  {
    v16h h0 = afrag(&Hs[m_lo * AS], 0, hi);
    v16h h1 = afrag(&Hs[m_lo * AS], 32, hi);
    v16h h2 = afrag(&Hs[m_lo * AS], 64, hi);
    v16h h3 = afrag(&Hs[m_lo * AS], 96, hi);
#pragma unroll
    for (int t = 0; t < 4; ++t) {
      v8f acc = {};
      acc = wmma_f16(h0, bfrag(w2p, t * 4 + 0, lane), acc);
      acc = wmma_f16(h1, bfrag(w2p, t * 4 + 1, lane), acc);
      acc = wmma_f16(h2, bfrag(w2p, t * 4 + 2, lane), acc);
      acc = wmma_f16(h3, bfrag(w2p, t * 4 + 3, lane), acc);
      const float bias = b2[t * 16 + m_lo];
#pragma unroll
      for (int r = 0; r < 8; ++r)
        atomicAdd(&W.X[(r + mb) * XS + t * 16 + m_lo], acc[r] + bias);
    }
  }
  __syncthreads();

  // ---- Stage 9: emit h_i = X[:, cls-token, :] (rows 0,4,8,12), coalesced ----
  {
    const int row_i = lane >> 3;
    const int part  = lane & 7;
    const int m     = row_i * 4;
    const int g     = G + row_i;
    if (g < Mtot) {
      v4f v0 = *(const v4f*)(&W.X[m * XS + part * 8]);
      v4f v1 = *(const v4f*)(&W.X[m * XS + part * 8 + 4]);
      *(v4f*)(out_h + (size_t)g * 64 + part * 8)     = v0;
      *(v4f*)(out_h + (size_t)g * 64 + part * 8 + 4) = v1;
    }
  }
}

extern "C" void kernel_launch(void* const* d_in, const int* in_sizes, int n_in,
                              void* d_out, int out_size, void* d_ws, size_t ws_size,
                              hipStream_t stream) {
  const float* z_m       = (const float*)d_in[0];
  const float* z_c       = (const float*)d_in[1];
  const float* z_t       = (const float*)d_in[2];
  const float* z_cls     = (const float*)d_in[3];
  const float* in_proj_w = (const float*)d_in[4];
  const float* in_proj_b = (const float*)d_in[5];
  const float* out_w     = (const float*)d_in[6];
  const float* out_b     = (const float*)d_in[7];
  const float* ln1_g     = (const float*)d_in[8];
  const float* ln1_b     = (const float*)d_in[9];
  const float* ln2_g     = (const float*)d_in[10];
  const float* ln2_b     = (const float*)d_in[11];
  const float* w1        = (const float*)d_in[12];
  const float* b1        = (const float*)d_in[13];
  const float* w2        = (const float*)d_in[14];
  const float* b2        = (const float*)d_in[15];
  float* out   = (float*)d_out;
  _Float16* ws = (_Float16*)d_ws;

  const int Mtot = in_sizes[0] / 64;  // B*N token groups

  pack_weights<<<(WTOTAL + 255) / 256, 256, 0, stream>>>(in_proj_w, out_w, w1, w2, ws);

  const int blocks = (Mtot + 7) / 8;  // 2 waves/block x 4 groups/wave
  mini_transformer_fused<<<blocks, 64, 0, stream>>>(
      z_m, z_c, z_t, z_cls, in_proj_b, out_b, ln1_g, ln1_b, ln2_g, ln2_b, b1, b2,
      ws + WQ_OFF, ws + WO_OFF, ws + W1_OFF, ws + W2_OFF,
      out, out + (size_t)Mtot * 64, Mtot);
}
